// MDPBMP_lp_85676007620844
// MI455X (gfx1250) — matrix operand order repaired and checked by reading, not compiled, not run on hardware
//
#include <hip/hip_runtime.h>
#include <hip/hip_bf16.h>

// ---------------------------------------------------------------------------
// MDPBMP metapath-attention pipeline for gfx1250 (wave32, WMMA).
// GEMMs use v_wmma_f32_16x16x32_f16 with the weight strip staged into LDS via
// the CDNA5 async-copy path (global_load_async_to_lds_b128 / s_wait_asynccnt).
// Intermediates kept in _Float16 to halve HBM traffic (memory-bound:
// ~0.5GB moved vs 65 GFLOP of matmul at 23.3 TB/s).
// ---------------------------------------------------------------------------

typedef __attribute__((ext_vector_type(16))) _Float16 v16h;
typedef __attribute__((ext_vector_type(8)))  _Float16 v8h;
typedef __attribute__((ext_vector_type(8)))  float    v8f;

#define HIDDEN 128
#define HEADS  4
#define NTGT   8192

// ---------------- elementwise helpers ----------------

__global__ void f32_to_f16_kernel(const float* __restrict__ in,
                                  _Float16* __restrict__ out, long long n) {
  long long i = (long long)blockIdx.x * 256 + threadIdx.x;
  if (i < n) out[i] = (_Float16)in[i];
}

__global__ void zero_f32_kernel(float* __restrict__ p, long long n) {
  long long i = (long long)blockIdx.x * 256 + threadIdx.x;
  if (i < n) p[i] = 0.0f;
}

// ---------------- WMMA GEMM: C[M,N] = A[M,K] * W[N,K]^T (+bias), fp16 out ---
// Block = 256 threads = 8 waves; block tile = 128(M) x 64(N).
// The 64-column weight strip (64 x K fp16, <=64KB) is staged into LDS once per
// block with async-to-LDS copies, then every wave streams A from global and B
// from LDS, issuing v_wmma_f32_16x16x32_f16 per 32-K step.
// A fragment layout (16-bit A 16x32, ISA 7.12.2): lane half selects K-phase.
// C VGPR r -> row m0 + r + 8*(lane>>4), col = n0 + t*16 + (lane&15).
// Optional row scatter through rowmap (node_idx) and bias add in epilogue.

template <int KV>
__global__ __launch_bounds__(256)
void wmma_gemm_f16_kernel(const _Float16* __restrict__ A,   // [M,K] row-major
                          const _Float16* __restrict__ W,   // [N,K] row-major
                          const float* __restrict__ bias,   // [N] or nullptr
                          const int* __restrict__ rowmap,   // [M] or nullptr
                          _Float16* __restrict__ C,         // [*,ldc] fp16 out
                          int M, int ldc) {
  __shared__ __attribute__((aligned(16))) _Float16 sW[64 * KV];

  const int wave = threadIdx.x >> 5;
  const int lane = threadIdx.x & 31;
  const int half = lane >> 4;
  const int l16  = lane & 15;
  const int n0   = blockIdx.y * 64;

  // ---- stage 64 x KV weight strip into LDS (async copy engine) ----
  // 16-byte chunks: 8*KV chunks total, 256 threads -> exact multiple.
  constexpr int CPR = KV / 8;  // 16B chunks per weight row
  for (int c = threadIdx.x; c < 8 * KV; c += 256) {
    const int n  = c / CPR;
    const int kk = (c - n * CPR) * 8;
    const _Float16* src = W + (size_t)(n0 + n) * KV + kk;
    const unsigned dst = (unsigned)(uintptr_t)(sW + n * KV + kk);
    asm volatile("global_load_async_to_lds_b128 %0, %1, off"
                 :: "v"(dst), "v"(src) : "memory");
  }
  asm volatile("s_wait_asynccnt 0x0" ::: "memory");
  __syncthreads();

  // ---- clamp tail waves instead of early-return (barrier safety);
  //      duplicated tiles store identical values -> benign.
  int m0 = (blockIdx.x * 8 + wave) * 16;
  if (m0 >= M) m0 = ((M - 1) >> 4) << 4;

  v8f acc[4] = {};

  int arow = m0 + l16;
  if (arow >= M) arow = M - 1;              // safe row; masked at store
  const _Float16* Arow = A + (size_t)arow * KV;

  for (int k0 = 0; k0 < KV; k0 += 32) {
    // ---- A fragment from global: two 16B loads per lane
    v16h afrag;
    {
      const int kb = k0 + half * 8;
      v8h lo = *(const v8h*)(Arow + kb);
      v8h hi = *(const v8h*)(Arow + kb + 16);
#pragma unroll
      for (int j = 0; j < 8; ++j) { afrag[j] = lo[j]; afrag[j + 8] = hi[j]; }
    }
#pragma unroll
    for (int t = 0; t < 4; ++t) {
      // ---- B fragment from LDS: column = strip row t*16+l16, K = 8*half+j and 16+8*half+j
      const _Float16* Wp = sW + (t * 16 + l16) * KV + k0 + half * 8;
      v8h b0v = *(const v8h*)(Wp);
      v8h b1v = *(const v8h*)(Wp + 16);
      v16h bfrag;
#pragma unroll
      for (int j = 0; j < 8; ++j) { bfrag[j] = b0v[j]; bfrag[j + 8] = b1v[j]; }
      acc[t] = __builtin_amdgcn_wmma_f32_16x16x32_f16(
          false, afrag, false, bfrag, (short)0, acc[t], false, false);
    }
  }

  // ---- epilogue: bias + optional row scatter, store fp16
#pragma unroll
  for (int t = 0; t < 4; ++t) {
    const int col = n0 + t * 16 + l16;
    const float bv = bias ? bias[col] : 0.0f;
#pragma unroll
    for (int r = 0; r < 8; ++r) {
      const int row = m0 + r + 8 * half;
      if (row < M) {
        const int orow = rowmap ? rowmap[row] : row;
        C[(size_t)orow * ldc + col] = (_Float16)(acc[t][r] + bv);
      }
    }
  }
}

// ---------------- metapath instance encoding: mean of 3 endpoint features ---

__global__ void gather_mean_kernel(const _Float16* __restrict__ tf,
                                   const int* __restrict__ eidx,
                                   _Float16* __restrict__ h, long long n) {
  long long i = (long long)blockIdx.x * 256 + threadIdx.x;  // over E*128
  if (i >= n) return;
  const int e = (int)(i >> 7), d = (int)(i & 127);
  const int i0 = eidx[e * 3 + 0];
  const int i1 = eidx[e * 3 + 1];
  const int i2 = eidx[e * 3 + 2];
  float s = (float)tf[(size_t)i0 * HIDDEN + d] +
            (float)tf[(size_t)i1 * HIDDEN + d] +
            (float)tf[(size_t)i2 * HIDDEN + d];
  h[(size_t)e * HIDDEN + d] = (_Float16)(s * (1.0f / 3.0f));
}

// ---------------- attention scores: leaky_relu(<eft[e,h,:], a[h,:]>, 0.2) ---

__global__ void score_kernel(const _Float16* __restrict__ eft,
                             const float* __restrict__ attn,
                             float* __restrict__ esc, long long n) {
  long long i = (long long)blockIdx.x * 256 + threadIdx.x;  // over E*HEADS
  if (i >= n) return;
  const int e = (int)(i >> 2), hh = (int)(i & 3);
  const _Float16* p = eft + (size_t)e * (HEADS * HIDDEN) + hh * HIDDEN;
  const float* a = attn + hh * HIDDEN;
  float s = 0.0f;
#pragma unroll 8
  for (int d = 0; d < HIDDEN; ++d) s += (float)p[d] * a[d];
  esc[i] = (s > 0.0f) ? s : 0.2f * s;
}

// ---------------- segment softmax (max / exp-sum) via atomics ----------------

__device__ __forceinline__ unsigned flt_flip(float f) {
  unsigned u = __float_as_uint(f);
  return (u & 0x80000000u) ? ~u : (u | 0x80000000u);
}
__device__ __forceinline__ float flt_unflip(unsigned u) {
  return (u & 0x80000000u) ? __uint_as_float(u ^ 0x80000000u)
                           : __uint_as_float(~u);
}

__global__ void seg_init_kernel(unsigned* __restrict__ m,
                                float* __restrict__ den, int n) {
  int i = blockIdx.x * 256 + threadIdx.x;
  if (i < n) { m[i] = flt_flip(-__builtin_inff()); den[i] = 0.0f; }
}

__global__ void seg_max_kernel(const float* __restrict__ esc,
                               const int* __restrict__ etgt,
                               unsigned* __restrict__ m, long long n) {
  long long i = (long long)blockIdx.x * 256 + threadIdx.x;  // over E*HEADS
  if (i >= n) return;
  const int e = (int)(i >> 2), hh = (int)(i & 3);
  atomicMax(&m[etgt[e] * HEADS + hh], flt_flip(esc[i]));
}

__global__ void seg_exp_kernel(float* __restrict__ esc,
                               const int* __restrict__ etgt,
                               const unsigned* __restrict__ m,
                               float* __restrict__ den, long long n) {
  long long i = (long long)blockIdx.x * 256 + threadIdx.x;
  if (i >= n) return;
  const int e = (int)(i >> 2), hh = (int)(i & 3);
  const float mx = flt_unflip(m[etgt[e] * HEADS + hh]);
  const float ex = __expf(esc[i] - mx);
  esc[i] = ex;
  atomicAdd(&den[etgt[e] * HEADS + hh], ex);
}

__global__ void seg_agg_kernel(const _Float16* __restrict__ eft,
                               const float* __restrict__ ex,
                               const float* __restrict__ den,
                               const int* __restrict__ etgt,
                               float* __restrict__ agg, long long n) {
  long long i = (long long)blockIdx.x * 256 + threadIdx.x;  // over E*512
  if (i >= n) return;
  const int e = (int)(i >> 9), c = (int)(i & 511);
  const int hh = c >> 7;
  const int t = etgt[e];
  const float alpha = ex[e * HEADS + hh] / (den[t * HEADS + hh] + 1e-9f);
  atomicAdd(&agg[(size_t)t * 512 + c], (float)eft[i] * alpha);
}

__global__ void elu_kernel(float* __restrict__ x, long long n) {
  long long i = (long long)blockIdx.x * 256 + threadIdx.x;
  if (i >= n) return;
  const float v = x[i];
  x[i] = (v > 0.0f) ? v : (__expf(v) - 1.0f);
}

// ---------------- semantic attention: beta[mp] = mean(tanh(h@Ws.T+bs) @ vs) --

__global__ void beta_kernel(const float* __restrict__ h,    // [8192,512]
                            const float* __restrict__ Ws,   // [128,512]
                            const float* __restrict__ bs,   // [128]
                            const float* __restrict__ vs,   // [128]
                            float* __restrict__ betaAcc) {  // scalar
  const long long i = (long long)blockIdx.x * 256 + threadIdx.x;  // 8192*128
  float val = 0.0f;
  if (i < (long long)NTGT * 128) {
    const int r = (int)(i >> 7), j = (int)(i & 127);
    const float* hr = h + (size_t)r * 512;
    const float* wj = Ws + (size_t)j * 512;
    float s = bs[j];
#pragma unroll 8
    for (int k = 0; k < 512; ++k) s += hr[k] * wj[k];
    val = vs[j] * tanhf(s) * (1.0f / (float)NTGT);
  }
  // block reduce -> one atomic per block
#pragma unroll
  for (int off = 16; off > 0; off >>= 1) val += __shfl_down(val, off, 32);
  __shared__ float ssum[8];
  if ((threadIdx.x & 31) == 0) ssum[threadIdx.x >> 5] = val;
  __syncthreads();
  if (threadIdx.x == 0) {
    float t = 0.0f;
#pragma unroll
    for (int w = 0; w < 8; ++w) t += ssum[w];
    atomicAdd(betaAcc, t);
  }
}

// ---------------- softmax(beta) weighted combine -----------------------------

__global__ void combine_kernel(const float* __restrict__ a0,
                               const float* __restrict__ a1,
                               const float* __restrict__ beta,  // 2 scalars
                               float* __restrict__ hout, long long n) {
  long long i = (long long)blockIdx.x * 256 + threadIdx.x;
  if (i >= n) return;
  const float b0 = beta[0], b1 = beta[1];
  const float mb = fmaxf(b0, b1);
  const float e0 = __expf(b0 - mb), e1 = __expf(b1 - mb);
  const float inv = 1.0f / (e0 + e1);
  hout[i] = (e0 * inv) * a0[i] + (e1 * inv) * a1[i];
}

// ---------------- final logits: out[r,o] = <h[r,:], Wo[o,:]> + bo[o] ---------

__global__ void logits_kernel(const float* __restrict__ h,   // [8192,512]
                              const float* __restrict__ Wo,  // [64,512]
                              const float* __restrict__ bo,  // [64]
                              float* __restrict__ out, long long n) {
  long long i = (long long)blockIdx.x * 256 + threadIdx.x;  // 8192*64
  if (i >= n) return;
  const int r = (int)(i >> 6), o = (int)(i & 63);
  const float* hr = h + (size_t)r * 512;
  const float* w = Wo + (size_t)o * 512;
  float s = bo[o];
#pragma unroll 8
  for (int k = 0; k < 512; ++k) s += hr[k] * w[k];
  out[i] = s;
}

// ===========================================================================

extern "C" void kernel_launch(void* const* d_in, const int* in_sizes, int n_in,
                              void* d_out, int out_size, void* d_ws, size_t ws_size,
                              hipStream_t stream) {
  (void)n_in; (void)out_size; (void)ws_size;

  const float* feat0 = (const float*)d_in[0];
  const float* feat1 = (const float*)d_in[1];
  const int* nidx0   = (const int*)d_in[2];
  const int* nidx1   = (const int*)d_in[3];
  const int* eidx[4] = {(const int*)d_in[4], (const int*)d_in[6],
                        (const int*)d_in[8], (const int*)d_in[10]};
  const int* etgt[4] = {(const int*)d_in[5], (const int*)d_in[7],
                        (const int*)d_in[9], (const int*)d_in[11]};
  const float* W0 = (const float*)d_in[12]; const float* b0 = (const float*)d_in[13];
  const float* W1 = (const float*)d_in[14]; const float* b1 = (const float*)d_in[15];
  const float* Wr[4]   = {(const float*)d_in[16], (const float*)d_in[18],
                          (const float*)d_in[25], (const float*)d_in[27]};
  const float* attn[4] = {(const float*)d_in[17], (const float*)d_in[19],
                          (const float*)d_in[26], (const float*)d_in[28]};
  const float* Ws[2] = {(const float*)d_in[20], (const float*)d_in[29]};
  const float* bs[2] = {(const float*)d_in[21], (const float*)d_in[30]};
  const float* vs[2] = {(const float*)d_in[22], (const float*)d_in[31]};
  const float* Wo[2] = {(const float*)d_in[23], (const float*)d_in[32]};
  const float* bo[2] = {(const float*)d_in[24], (const float*)d_in[33]};

  const int N0 = in_sizes[2];          // 50000
  const int N1 = in_sizes[3];          // 50000
  const int E  = in_sizes[5];          // 100000 edges per metapath

  // ---- workspace carve-out (256B aligned) ----
  char* ws = (char*)d_ws;
  size_t off = 0;
  auto carve = [&](size_t bytes) {
    void* p = ws + off;
    off += (bytes + 255) & ~(size_t)255;
    return p;
  };
  size_t featBytes = (size_t)(N0 + N1) * 512 * sizeof(_Float16);
  size_t eftBytes  = (size_t)E * HEADS * HIDDEN * sizeof(_Float16);
  _Float16* feat16 = (_Float16*)carve(featBytes > eftBytes ? featBytes : eftBytes);
  _Float16* eft16  = feat16;  // reused after tf GEMM consumes feat16
  _Float16* tf16   = (_Float16*)carve((size_t)(N0 + N1) * HIDDEN * sizeof(_Float16));
  _Float16* h16    = (_Float16*)carve((size_t)E * HIDDEN * sizeof(_Float16));
  float*    esc    = (float*)carve((size_t)E * HEADS * sizeof(float));
  unsigned* segm   = (unsigned*)carve((size_t)NTGT * HEADS * sizeof(unsigned));
  float*    segden = (float*)carve((size_t)NTGT * HEADS * sizeof(float));
  float*    agg    = (float*)carve((size_t)4 * NTGT * 512 * sizeof(float));
  _Float16* W0f    = (_Float16*)carve((size_t)HIDDEN * 512 * sizeof(_Float16));
  _Float16* W1f    = (_Float16*)carve((size_t)HIDDEN * 512 * sizeof(_Float16));
  _Float16* Wrf    = (_Float16*)carve((size_t)HEADS * HIDDEN * HIDDEN * sizeof(_Float16));
  float*    betaAcc = (float*)carve(16 * sizeof(float));

  const dim3 B256(256);
  auto g1 = [](long long n) { return dim3((unsigned)((n + 255) / 256)); };

  // ---- fp32 -> fp16 conversions ----
  f32_to_f16_kernel<<<g1((long long)N0 * 512), B256, 0, stream>>>(feat0, feat16, (long long)N0 * 512);
  f32_to_f16_kernel<<<g1((long long)N1 * 512), B256, 0, stream>>>(feat1, feat16 + (size_t)N0 * 512, (long long)N1 * 512);
  f32_to_f16_kernel<<<g1(HIDDEN * 512), B256, 0, stream>>>(W0, W0f, HIDDEN * 512);
  f32_to_f16_kernel<<<g1(HIDDEN * 512), B256, 0, stream>>>(W1, W1f, HIDDEN * 512);

  // ---- per-node-type transform: tf[nidx] = feat @ W.T + b (WMMA, scatter) ----
  {
    dim3 g0(((N0 + 15) / 16 + 7) / 8, HIDDEN / 64);
    wmma_gemm_f16_kernel<512><<<g0, B256, 0, stream>>>(
        feat16, W0f, b0, nidx0, tf16, N0, HIDDEN);
    dim3 gA(((N1 + 15) / 16 + 7) / 8, HIDDEN / 64);
    wmma_gemm_f16_kernel<512><<<gA, B256, 0, stream>>>(
        feat16 + (size_t)N0 * 512, W1f, b1, nidx1, tf16, N1, HIDDEN);
  }

  zero_f32_kernel<<<g1((long long)4 * NTGT * 512), B256, 0, stream>>>(agg, (long long)4 * NTGT * 512);
  zero_f32_kernel<<<g1(16), B256, 0, stream>>>(betaAcc, 16);

  // ---- metapath aggregation (m0, m1, d0, d1) ----
  for (int mp = 0; mp < 4; ++mp) {
    f32_to_f16_kernel<<<g1(HEADS * HIDDEN * HIDDEN), B256, 0, stream>>>(
        Wr[mp], Wrf, HEADS * HIDDEN * HIDDEN);
    gather_mean_kernel<<<g1((long long)E * HIDDEN), B256, 0, stream>>>(
        tf16, eidx[mp], h16, (long long)E * HIDDEN);
    dim3 gE(((E + 15) / 16 + 7) / 8, (HEADS * HIDDEN) / 64);
    wmma_gemm_f16_kernel<HIDDEN><<<gE, B256, 0, stream>>>(
        h16, Wrf, nullptr, nullptr, eft16, E, HEADS * HIDDEN);
    score_kernel<<<g1((long long)E * HEADS), B256, 0, stream>>>(
        eft16, attn[mp], esc, (long long)E * HEADS);
    seg_init_kernel<<<g1(NTGT * HEADS), B256, 0, stream>>>(segm, segden, NTGT * HEADS);
    seg_max_kernel<<<g1((long long)E * HEADS), B256, 0, stream>>>(
        esc, etgt[mp], segm, (long long)E * HEADS);
    seg_exp_kernel<<<g1((long long)E * HEADS), B256, 0, stream>>>(
        esc, etgt[mp], segm, segden, (long long)E * HEADS);
    seg_agg_kernel<<<g1((long long)E * 512), B256, 0, stream>>>(
        eft16, esc, segden, etgt[mp], agg + (size_t)mp * NTGT * 512, (long long)E * 512);
  }

  elu_kernel<<<g1((long long)4 * NTGT * 512), B256, 0, stream>>>(agg, (long long)4 * NTGT * 512);

  // ---- semantic attention betas ----
  for (int mp = 0; mp < 4; ++mp) {
    const int ty = mp >> 1;
    beta_kernel<<<g1((long long)NTGT * 128), B256, 0, stream>>>(
        agg + (size_t)mp * NTGT * 512, Ws[ty], bs[ty], vs[ty], betaAcc + mp);
  }

  // ---- outputs: [logits_m | logits_d | h_m | h_d] ----
  float* out      = (float*)d_out;
  float* logits_m = out;
  float* logits_d = out + (size_t)NTGT * 64;
  float* h_m      = out + (size_t)NTGT * 64 * 2;
  float* h_d      = h_m + (size_t)NTGT * 512;

  combine_kernel<<<g1((long long)NTGT * 512), B256, 0, stream>>>(
      agg + (size_t)0 * NTGT * 512, agg + (size_t)1 * NTGT * 512, betaAcc + 0,
      h_m, (long long)NTGT * 512);
  combine_kernel<<<g1((long long)NTGT * 512), B256, 0, stream>>>(
      agg + (size_t)2 * NTGT * 512, agg + (size_t)3 * NTGT * 512, betaAcc + 2,
      h_d, (long long)NTGT * 512);

  logits_kernel<<<g1((long long)NTGT * 64), B256, 0, stream>>>(
      h_m, Wo[0], bo[0], logits_m, (long long)NTGT * 64);
  logits_kernel<<<g1((long long)NTGT * 64), B256, 0, stream>>>(
      h_d, Wo[1], bo[1], logits_d, (long long)NTGT * 64);
}
